// MultiheadAttention_60636348285324
// MI455X (gfx1250) — matrix operand order repaired and checked by reading, not compiled, not run on hardware
//
#include <hip/hip_runtime.h>

#define Bsz  4
#define Sdim 2048
#define Ddim 1024
#define Hn   16
#define DHd  64

typedef __attribute__((ext_vector_type(16))) __bf16       v16bf;
typedef __attribute__((ext_vector_type(8)))  float        v8f;
typedef __attribute__((ext_vector_type(4)))  unsigned int v4u;
typedef __attribute__((ext_vector_type(8)))  int          v8i;
typedef __attribute__((ext_vector_type(4)))  int          v4i;

struct __attribute__((aligned(16))) u32x4 { unsigned int x, y, z, w; };
struct u32x8 { u32x4 a, b; };

// native bf16 converts (v_cvt_* on gfx1250; round-to-nearest-even)
__device__ __forceinline__ unsigned short f2bf(float f) {
  return __builtin_bit_cast(unsigned short, (__bf16)f);
}
__device__ __forceinline__ float bf2f(unsigned short s) {
  return (float)__builtin_bit_cast(__bf16, s);
}
__device__ __forceinline__ v8f vzero8() {
  v8f z = {0.f, 0.f, 0.f, 0.f, 0.f, 0.f, 0.f, 0.f};
  return z;
}

// async DMA: global -> LDS, 16 bytes per lane (ASYNCcnt)
__device__ __forceinline__ void async_copy_b128(const void* gsrc, const void* ldst) {
  unsigned loff = (unsigned)(uintptr_t)ldst;   // low 32 bits of generic addr == LDS byte offset
  asm volatile("global_load_async_to_lds_b128 %0, %1, off"
               :: "v"(loff), "v"(gsrc) : "memory");
}
__device__ __forceinline__ void wait_async() {
  asm volatile("s_wait_asynccnt 0x0" ::: "memory");
}

// Tensor Data Mover: load a 64x64 2D bf16 tile (row pitch = stride0 elements) into LDS,
// padding the destination with 4 DWORDs (8 halfwords) after every 32 DWORDs (64 halfwords)
// -> matches the [64][72] LDS tiles. Tracked by TENSORcnt. Issue from ONE wave only.
__device__ __forceinline__ void tdm_load_tile_2d(unsigned lds_addr, const void* gaddr,
                                                 unsigned tdim0, unsigned tdim1, unsigned stride0) {
  unsigned long long ga = (unsigned long long)(uintptr_t)gaddr;
  v4u g0;
  g0[0] = 1u;                                              // count=1 (valid user descriptor)
  g0[1] = lds_addr;                                        // D#.lds_addr
  g0[2] = (unsigned)(ga & 0xFFFFFFFFu);                    // global_addr[31:0]
  g0[3] = (unsigned)((ga >> 32) & 0x01FFFFFFu) | 0x80000000u;  // global_addr[56:32] | type=2
  v8i g1;
  g1[0] = (int)((1u << 16) |                               // data_size = 2 bytes
                (1u << 20) |                               // pad_enable
                (4u << 22) |                               // pad_interval: 32 DWORDs
                (3u << 25));                               // pad_amount: 4 DWORDs
  g1[1] = (int)((tdim0 & 0xFFFFu) << 16);                  // tensor_dim0[15:0]
  g1[2] = (int)((tdim0 >> 16) | ((tdim1 & 0xFFFFu) << 16));
  g1[3] = (int)((tdim1 >> 16) | (64u << 16));              // tile_dim0 = 64
  g1[4] = (int)64u;                                        // tile_dim1 = 64, tile_dim2 = 0
  g1[5] = (int)stride0;                                    // tensor_dim0_stride (row pitch)
  g1[6] = 0;
  g1[7] = 0;
  v4i z4 = {0, 0, 0, 0};
  v8i z8 = {0, 0, 0, 0, 0, 0, 0, 0};
  __builtin_amdgcn_tensor_load_to_lds(g0, g1, z4, z4, z8, 0);  // 6-arg toolchain form
}

// A fragment (16x32 bf16, row-major tile): lane m, K {kb..kb+7, kb+16..kb+23}, kb=8*(lane>=16)
template<int LDA>
__device__ __forceinline__ v16bf load_frag_a(const unsigned short (*A)[LDA], int mbase, int kofs, int lane) {
  const int m  = mbase + (lane & 15);
  const int kb = kofs + ((lane >> 4) << 3);
  u32x8 t;
  t.a = *(const u32x4*)&A[m][kb];
  t.b = *(const u32x4*)&A[m][kb + 16];
  return __builtin_bit_cast(v16bf, t);
}

// B fragment (32x16 bf16) from a COLUMN-MAJOR tile Bc[n][k]: lane n, K {kb..kb+7, kb+16..kb+23}, kb=kofs+8*(lane>=16)
template<int LDB>
__device__ __forceinline__ v16bf load_frag_b_cm(const unsigned short (*Bc)[LDB], int nbase, int kofs, int lane) {
  const int n  = nbase + (lane & 15);
  const int kb = kofs + ((lane >> 4) << 3);
  u32x8 t;
  t.a = *(const u32x4*)&Bc[n][kb];
  t.b = *(const u32x4*)&Bc[n][kb + 16];
  return __builtin_bit_cast(v16bf, t);
}

// -------- Projection GEMM: Y = X @ W^T  (X: [8192,1024], W: [1024,1024] row-major) --------
// Block tile 128(M) x 64(N), K-step 32. 8 waves, wave w owns rows w*16..w*16+15, all 64 cols.
// mode 0: Y -> bf16 [B,H,S,DH]   (Q and K; Q uses oscale = 1/sqrt(DH))
// mode 1: Y -> bf16 [B,H,DH,S]   (V transposed, packed 16B stores along s)
// mode 3: Y -> fp32 row-major    (final output projection)
template<bool XF32>
__global__ __launch_bounds__(256)
void proj_gemm_kernel(const void* __restrict__ Xv, const float* __restrict__ W,
                      unsigned short* __restrict__ Yb, float* __restrict__ Yf,
                      int mode, float oscale) {
  __shared__ __attribute__((aligned(16))) unsigned short As[128][48];
  __shared__ __attribute__((aligned(16))) unsigned short Bc[64][48];

  const int tid  = threadIdx.x;
  const int lane = tid & 31;
  const int wave = tid >> 5;
  const int m0   = blockIdx.y * 128;
  const int n0   = blockIdx.x * 64;

  const int ar = tid >> 1, ak = (tid & 1) * 16;
  const int br = tid >> 2, bk = (tid & 3) * 8;

  v8f acc[4] = { vzero8(), vzero8(), vzero8(), vzero8() };

  for (int k0 = 0; k0 < Ddim; k0 += 32) {
    if (k0 + 32 < Ddim) {                       // -> global_prefetch_b8
      if (XF32) __builtin_prefetch((const float*)Xv + (size_t)(m0 + ar) * Ddim + k0 + 32 + ak, 0, 1);
      else      __builtin_prefetch((const unsigned short*)Xv + (size_t)(m0 + ar) * Ddim + k0 + 32 + ak, 0, 1);
      __builtin_prefetch(W + (size_t)(n0 + br) * Ddim + k0 + 32 + bk, 0, 1);
    }
    if (XF32) {
      const float* xs = (const float*)Xv + (size_t)(m0 + ar) * Ddim + k0 + ak;
      union { u32x4 q[2]; unsigned short s[16]; } t;
#pragma unroll
      for (int j = 0; j < 16; ++j) t.s[j] = f2bf(xs[j]);
      ((u32x4*)&As[ar][ak])[0] = t.q[0];
      ((u32x4*)&As[ar][ak])[1] = t.q[1];
    } else {
      const u32x4* xs = (const u32x4*)((const unsigned short*)Xv + (size_t)(m0 + ar) * Ddim + k0 + ak);
      ((u32x4*)&As[ar][ak])[0] = xs[0];
      ((u32x4*)&As[ar][ak])[1] = xs[1];
    }
    {  // Bc[n][k] = W[n0+n][k0+k]  (column-major B tile == plain row copy of W)
      const float* wsrc = W + (size_t)(n0 + br) * Ddim + k0 + bk;
      union { u32x4 q; unsigned short s[8]; } t;
#pragma unroll
      for (int j = 0; j < 8; ++j) t.s[j] = f2bf(wsrc[j]);
      *(u32x4*)&Bc[br][bk] = t.q;
    }
    __syncthreads();

    v16bf a = load_frag_a<48>(As, wave * 16, 0, lane);
#pragma unroll
    for (int j = 0; j < 4; ++j) {
      v16bf b = load_frag_b_cm<48>(Bc, j * 16, 0, lane);
      acc[j] = __builtin_amdgcn_wmma_f32_16x16x32_bf16(false, a, false, b, (short)0, acc[j], false, false);
    }
    __syncthreads();
  }

  const int mb = m0 + wave * 16 + ((lane >> 4) << 3);
#pragma unroll
  for (int j = 0; j < 4; ++j) {
    const int ng = n0 + j * 16 + (lane & 15);
    if (mode == 3) {
#pragma unroll
      for (int r = 0; r < 8; ++r) Yf[(size_t)(mb + r) * Ddim + ng] = acc[j][r] * oscale;
    } else if (mode == 1) {      // V^T [B,H,DH,S]: 8 consecutive s per lane -> one 16B store
      const int b = mb >> 11, s = mb & (Sdim - 1);
      const int h = ng >> 6,  dh = ng & (DHd - 1);
      union { u32x4 q; unsigned short us_[8]; } t;
#pragma unroll
      for (int r = 0; r < 8; ++r) t.us_[r] = f2bf(acc[j][r] * oscale);
      *(u32x4*)(Yb + (((size_t)b * Hn + h) * DHd + dh) * Sdim + s) = t.q;
    } else {                     // Q/K [B,H,S,DH]
      const int h = ng >> 6, dh = ng & (DHd - 1);
#pragma unroll
      for (int r = 0; r < 8; ++r) {
        const int mg = mb + r;
        const int b = mg >> 11, s = mg & (Sdim - 1);
        Yb[(((size_t)b * Hn + h) * Sdim + s) * DHd + dh] = f2bf(acc[j][r] * oscale);
      }
    }
  }
}

// -------- Flash attention: per (b,h), 64 query rows per block, causal --------
// Q staged once via async DMA (ASYNCcnt). K/V tiles staged by the Tensor Data Mover
// (TENSORcnt), double-buffered so the DMA of tile kt+1 overlaps compute on tile kt.
__global__ __launch_bounds__(256)
void attention_kernel(const unsigned short* __restrict__ Qh,   // [B,H,S,DH] bf16 (pre-scaled)
                      const unsigned short* __restrict__ Kh,   // [B,H,S,DH] bf16
                      const unsigned short* __restrict__ VhT,  // [B,H,DH,S] bf16
                      unsigned short* __restrict__ Ah) {       // [B,S,D] bf16
  __shared__ __attribute__((aligned(16))) unsigned short Qs[64][72];
  __shared__ __attribute__((aligned(16))) unsigned short Ks[2][64][72];   // [k][dh]
  __shared__ __attribute__((aligned(16))) unsigned short VsT[2][64][72];  // [dh][k]
  __shared__ __attribute__((aligned(16))) unsigned short Ps[64][72];
  __shared__ float Sc[64][65];
  __shared__ float mrow[64], lsum[64], arow[64];
  __shared__ float red[64][4];

  const int tid  = threadIdx.x;
  const int lane = tid & 31;
  const int wave = tid >> 5;
  const int wm = wave >> 1, wn = wave & 1;
  const int qt = blockIdx.x;
  const int bh = blockIdx.y;
  const int q0 = qt * 64;

  const int lrow = tid >> 2;        // 0..63
  const int lc   = (tid & 3) * 16;  // 0,16,32,48

  auto stage_kv_tdm = [&](int kt, int buf) {   // call from wave 0 only (uniform)
    const int k0 = kt * 64;
    const unsigned short* kg = Kh  + ((size_t)bh * Sdim + k0) * DHd;
    tdm_load_tile_2d((unsigned)(uintptr_t)&Ks[buf][0][0], kg,
                     /*tdim0=*/DHd, /*tdim1=*/(unsigned)Sdim, /*stride0=*/DHd);
    const unsigned short* vg = VhT + (size_t)bh * DHd * Sdim + k0;
    tdm_load_tile_2d((unsigned)(uintptr_t)&VsT[buf][0][0], vg,
                     /*tdim0=*/(unsigned)Sdim, /*tdim1=*/DHd, /*stride0=*/(unsigned)Sdim);
  };

  // stage Q tile via async DMA (already scaled by 1/sqrt(DH) in the projection)
  {
    const unsigned short* src = Qh + ((size_t)bh * Sdim + (q0 + lrow)) * DHd + lc;
    async_copy_b128(src,     &Qs[lrow][lc]);
    async_copy_b128(src + 8, &Qs[lrow][lc + 8]);
  }
  if (tid < 64) { mrow[tid] = -3.0e38f; lsum[tid] = 0.f; }

  if (wave == 0) {                     // prologue TDM for tile 0
    stage_kv_tdm(0, 0);
    __builtin_amdgcn_s_wait_tensorcnt(0);
  }
  wait_async();
  __syncthreads();

  v8f o[2] = { vzero8(), vzero8() };

  for (int kt = 0; kt <= qt; ++kt) {
    const int cur = kt & 1;
    const int k0  = kt * 64;
    if (wave == 0 && kt + 1 <= qt) stage_kv_tdm(kt + 1, cur ^ 1);  // overlap DMA with compute

    // S = Q K^T : wave owns a 16x32 region of the 64x64 score tile
    v8f sc[2] = { vzero8(), vzero8() };
#pragma unroll
    for (int ks = 0; ks < 64; ks += 32) {
      v16bf a = load_frag_a<72>(Qs, wm * 16, ks, lane);
#pragma unroll
      for (int j = 0; j < 2; ++j) {
        v16bf b = load_frag_b_cm<72>(Ks[cur], wn * 32 + j * 16, ks, lane);
        sc[j] = __builtin_amdgcn_wmma_f32_16x16x32_bf16(false, a, false, b, (short)0, sc[j], false, false);
      }
    }
#pragma unroll
    for (int j = 0; j < 2; ++j)
#pragma unroll
      for (int r = 0; r < 8; ++r)
        Sc[wm * 16 + r + ((lane >> 4) << 3)][wn * 32 + j * 16 + (lane & 15)] = sc[j][r];
    __syncthreads();

    // online softmax across this 64-wide k-tile (4 threads per q row)
    const int qg = q0 + lrow;
    float tmax = -3.0e38f;
#pragma unroll
    for (int i = 0; i < 16; ++i) {
      const int c = lc + i;
      float s = Sc[lrow][c];
      if (k0 + c > qg) s = -3.0e38f;       // causal mask
      tmax = fmaxf(tmax, s);
    }
    red[lrow][tid & 3] = tmax;
    __syncthreads();
    const float tilemax = fmaxf(fmaxf(red[lrow][0], red[lrow][1]),
                                fmaxf(red[lrow][2], red[lrow][3]));
    const float mold  = mrow[lrow];
    const float mnew  = fmaxf(mold, tilemax);
    const float alpha = __expf(mold - mnew);
    float psum = 0.f;
#pragma unroll
    for (int i = 0; i < 16; ++i) {
      const int c = lc + i;
      float p = 0.f;
      if (k0 + c <= qg) p = __expf(Sc[lrow][c] - mnew);
      Ps[lrow][c] = f2bf(p);
      psum += p;
    }
    __syncthreads();
    red[lrow][tid & 3] = psum;
    __syncthreads();
    if ((tid & 3) == 0) {
      lsum[lrow] = alpha * lsum[lrow] + red[lrow][0] + red[lrow][1] + red[lrow][2] + red[lrow][3];
      mrow[lrow] = mnew;
      arow[lrow] = alpha;
    }
    __syncthreads();

    // O = alpha * O + P @ V
#pragma unroll
    for (int j = 0; j < 2; ++j)
#pragma unroll
      for (int r = 0; r < 8; ++r)
        o[j][r] *= arow[wm * 16 + r + ((lane >> 4) << 3)];
#pragma unroll
    for (int ks = 0; ks < 64; ks += 32) {
      v16bf a = load_frag_a<72>(Ps, wm * 16, ks, lane);
#pragma unroll
      for (int j = 0; j < 2; ++j) {
        v16bf b = load_frag_b_cm<72>(VsT[cur], wn * 32 + j * 16, ks, lane);
        o[j] = __builtin_amdgcn_wmma_f32_16x16x32_bf16(false, a, false, b, (short)0, o[j], false, false);
      }
    }
    if (wave == 0) __builtin_amdgcn_s_wait_tensorcnt(0);  // next tile resident before reuse
    __syncthreads();
  }

  // normalize and write Ah[b][s][h*DH + dh]
  const int b = bh >> 4, h = bh & 15;
#pragma unroll
  for (int j = 0; j < 2; ++j)
#pragma unroll
    for (int r = 0; r < 8; ++r) {
      const int m  = wm * 16 + r + ((lane >> 4) << 3);
      const int dh = wn * 32 + j * 16 + (lane & 15);
      const float v = o[j][r] / lsum[m];
      Ah[((size_t)b * Sdim + (q0 + m)) * Ddim + h * DHd + dh] = f2bf(v);
    }
}

extern "C" void kernel_launch(void* const* d_in, const int* in_sizes, int n_in,
                              void* d_out, int out_size, void* d_ws, size_t ws_size,
                              hipStream_t stream) {
  (void)in_sizes; (void)n_in; (void)out_size; (void)ws_size;
  const float* q  = (const float*)d_in[0];
  const float* k  = (const float*)d_in[1];
  const float* v  = (const float*)d_in[2];
  // d_in[3] is the causal tril mask; applied analytically (k <= q) in-kernel.
  const float* wq = (const float*)d_in[4];
  const float* wk = (const float*)d_in[5];
  const float* wv = (const float*)d_in[6];
  const float* wo = (const float*)d_in[7];
  float* out = (float*)d_out;

  const size_t NTOK = (size_t)Bsz * Sdim * Ddim;           // 8M elements
  unsigned short* Qh  = (unsigned short*)d_ws;             // 16 MB  [B,H,S,DH] (pre-scaled)
  unsigned short* Kh  = Qh + NTOK;                         // 16 MB  [B,H,S,DH]
  unsigned short* VhT = Kh + NTOK;                         // 16 MB  [B,H,DH,S]
  unsigned short* Ah  = VhT + NTOK;                        // 16 MB  [B,S,D]

  dim3 gproj(Ddim / 64, (Bsz * Sdim) / 128);               // (16, 64)
  proj_gemm_kernel<true><<<gproj, 256, 0, stream>>>(q, wq, Qh,  nullptr, 0, 0.125f);  // 1/sqrt(64)
  proj_gemm_kernel<true><<<gproj, 256, 0, stream>>>(k, wk, Kh,  nullptr, 0, 1.0f);
  proj_gemm_kernel<true><<<gproj, 256, 0, stream>>>(v, wv, VhT, nullptr, 1, 1.0f);

  dim3 gattn(Sdim / 64, Bsz * Hn);                         // (32, 64)
  attention_kernel<<<gattn, 256, 0, stream>>>(Qh, Kh, VhT, Ah);

  proj_gemm_kernel<false><<<gproj, 256, 0, stream>>>(Ah, wo, nullptr, out, 3, 1.0f);
}
